// MultiScaleAttention_22368189677630
// MI455X (gfx1250) — matrix operand order repaired and checked by reading, not compiled, not run on hardware
//
#include <hip/hip_runtime.h>
#include <hip/hip_bf16.h>
#include <math.h>

// ---------------- problem constants (match reference) ----------------
constexpr int CB   = 2;                   // batch
constexpr int CHD  = 96;                  // head dim (NH=1)
constexpr int CT   = 8;
constexpr int CH   = 28;
constexpr int CW   = 28;
constexpr int CHW1 = 1 + CH * CW;         // 785
constexpr int CN   = CT * CHW1;           // 6280 tokens per batch
constexpr int CM   = CB * CN;             // 12560 rows for GEMMs (785*16)
constexpr int CQT  = (CN + 15) / 16;      // 393 query tiles per batch
constexpr int CKT  = (CN + 31) / 32;      // 197 key tiles per batch
constexpr int CNP  = CKT * 32;            // 6304 padded keys (for V^T)

typedef __attribute__((ext_vector_type(16))) _Float16 v16h;
typedef __attribute__((ext_vector_type(8)))  _Float16 v8h;
typedef __attribute__((ext_vector_type(8)))  float    v8f;
typedef __attribute__((ext_vector_type(4)))  float    v4f;

// ---------------- WMMA fragment loaders (32-bit index math) ----------------
// A-operand 16x32 (MxK), fp32 source, rows guaranteed in range.
__device__ inline v16h load_a_f32(const float* A, int lda, int row0, int k0) {
  int lane = threadIdx.x & 31;
  int r = row0 + (lane & 15);
  int kb = k0 + ((lane >> 4) << 3);
  const float* p = A + (r * lda + kb);
  v4f a0 = *(const v4f*)(p);
  v4f a1 = *(const v4f*)(p + 4);
  v4f a2 = *(const v4f*)(p + 16);
  v4f a3 = *(const v4f*)(p + 20);
  v16h f;
#pragma unroll
  for (int i = 0; i < 4; ++i) {
    f[i]      = (_Float16)a0[i];
    f[4 + i]  = (_Float16)a1[i];
    f[8 + i]  = (_Float16)a2[i];
    f[12 + i] = (_Float16)a3[i];
  }
  return f;
}

// A-operand 16x32 from contiguous f16 rows (clamped: last query tile partial).
__device__ inline v16h load_a_f16(const _Float16* A, int lda, int row0, int k0,
                                  int rowMax) {
  int lane = threadIdx.x & 31;
  int r = row0 + (lane & 15);
  if (r > rowMax) r = rowMax;
  int kb = k0 + ((lane >> 4) << 3);
  const _Float16* p = A + (r * lda + kb);
  v8h lo = *(const v8h*)p;
  v8h hi = *(const v8h*)(p + 16);
  v16h f;
#pragma unroll
  for (int i = 0; i < 8; ++i) { f[i] = lo[i]; f[8 + i] = hi[i]; }
  return f;
}

// B-operand 32x16 (KxN) from row-major fp32 weights (K multiple of 32).
__device__ inline v16h load_b_rows_f32(const float* W, int ldw, int k0, int n0) {
  int lane = threadIdx.x & 31;
  int n = n0 + (lane & 15);
  int kb = k0 + ((lane >> 4) << 3);
  const float* p = W + (kb * ldw + n);
  v16h f;
#pragma unroll
  for (int e = 0; e < 16; ++e) f[e] = (_Float16)p[(e < 8 ? e : e + 8) * ldw];
  return f;
}

// B-operand for Q@K^T from the per-wave LDS K tile (keys x 96 f16).
__device__ inline v16h load_b_kf16_lds(const _Float16* kt, int key0, int c0) {
  int lane = threadIdx.x & 31;
  int lkey = key0 + (lane & 15);
  int kb = c0 + ((lane >> 4) << 3);
  const _Float16* p = kt + (lkey * CHD + kb);
  v8h lo = *(const v8h*)p;
  v8h hi = *(const v8h*)(p + 16);
  v16h f;
#pragma unroll
  for (int i = 0; i < 8; ++i) { f[i] = lo[i]; f[8 + i] = hi[i]; }
  return f;
}

// B-operand for P@V from transposed f16 V^T (96 x CNP): keys contiguous.
__device__ inline v16h load_b_vt_f16(const _Float16* vtB, int c0, int k0) {
  int lane = threadIdx.x & 31;
  int c = c0 + (lane & 15);
  int kb = k0 + ((lane >> 4) << 3);
  const _Float16* p = vtB + (c * CNP + kb);
  v8h lo = *(const v8h*)p;
  v8h hi = *(const v8h*)(p + 16);
  v16h f;
#pragma unroll
  for (int i = 0; i < 8; ++i) { f[i] = lo[i]; f[8 + i] = hi[i]; }
  return f;
}

// D layout store, no guards (M multiple of 16 in GEMM path).
__device__ inline void store_d(float* C, int ldc, int m0, int n0, v8f d) {
  int lane = threadIdx.x & 31;
  int n = n0 + (lane & 15);
  int mb = m0 + ((lane >> 4) << 3);
  float* p = C + (mb * ldc + n);
#pragma unroll
  for (int i = 0; i < 8; ++i) p[i * ldc] = d[i];
}

// ---------------- generic WMMA GEMM: C = A(MxK) * W(KxN) + bias ----------------
__global__ __launch_bounds__(128) void wmma_gemm_bias(const float* __restrict__ A,
                                                      const float* __restrict__ W,
                                                      const float* __restrict__ bias,
                                                      float* __restrict__ C,
                                                      int M, int N, int K) {
  int wave = blockIdx.x * (blockDim.x >> 5) + (threadIdx.x >> 5);
  int tilesN = N >> 4;
  int tm = wave / tilesN;
  int tn = wave - tm * tilesN;
  if (tm * 16 >= M) return;
  int lane = threadIdx.x & 31;
  float bv = bias[tn * 16 + (lane & 15)];
  v8f acc;
#pragma unroll
  for (int i = 0; i < 8; ++i) acc[i] = bv;
  for (int k0 = 0; k0 < K; k0 += 32) {
    v16h a = load_a_f32(A, K, tm * 16, k0);
    v16h b = load_b_rows_f32(W, N, k0, tn * 16);
    acc = __builtin_amdgcn_wmma_f32_16x16x32_f16(false, a, false, b,
                                                 (short)0, acc, false, false);
  }
  store_d(C, N, tm * 16, tn * 16, acc);
}

// ---------------- depthwise 3x3 pooling (cls passthrough) ----------------
__global__ void pool_conv(const float* __restrict__ src, int src_ld,
                          const float* __restrict__ w33,
                          float* __restrict__ dst, int total) {
  int idx = blockIdx.x * blockDim.x + threadIdx.x;
  if (idx >= total) return;
  int c = idx % CHD;
  int tok = idx / CHD;              // b*CN + n
  int n = tok % CN;
  int j = n % CHW1;
  float out;
  if (j == 0) {
    out = src[tok * src_ld + c];
  } else {
    int p = j - 1;
    int h = p / CW, w = p % CW;
    const float* wc = w33 + c * 9;
    out = 0.f;
#pragma unroll
    for (int dh = -1; dh <= 1; ++dh)
#pragma unroll
      for (int dw = -1; dw <= 1; ++dw) {
        int hh = h + dh, ww = w + dw;
        if (hh >= 0 && hh < CH && ww >= 0 && ww < CW) {
          int tok2 = tok - j + 1 + hh * CW + ww;
          out += src[tok2 * src_ld + c] * wc[(dh + 1) * 3 + (dw + 1)];
        }
      }
  }
  dst[tok * CHD + c] = out;
}

// ---------------- per-token LayerNorm (in place) ----------------
__global__ void layernorm_inplace(float* __restrict__ x,
                                  const float* __restrict__ g,
                                  const float* __restrict__ bb, int ntok) {
  int tok = blockIdx.x * blockDim.x + threadIdx.x;
  if (tok >= ntok) return;
  float* p = x + tok * CHD;
  float m = 0.f;
  for (int c = 0; c < CHD; ++c) m += p[c];
  m *= (1.0f / CHD);
  float v = 0.f;
  for (int c = 0; c < CHD; ++c) { float d = p[c] - m; v += d * d; }
  v *= (1.0f / CHD);
  float inv = rsqrtf(v + 1e-6f);
  for (int c = 0; c < CHD; ++c) p[c] = (p[c] - m) * inv * g[c] + bb[c];
}

// ---------------- f16 repack: qh = scale*q, kh = k, vt = V^T (zero-padded) ----
__global__ void cvt_qk(const float* __restrict__ pq, const float* __restrict__ pk,
                       _Float16* __restrict__ qh, _Float16* __restrict__ kh,
                       float scale, int total) {
  int idx = blockIdx.x * blockDim.x + threadIdx.x;
  if (idx >= total) return;
  qh[idx] = (_Float16)(pq[idx] * scale);
  kh[idx] = (_Float16)pk[idx];
}

__global__ void cvt_vt(const float* __restrict__ pv, _Float16* __restrict__ vt,
                       int total) {
  int idx = blockIdx.x * blockDim.x + threadIdx.x;
  if (idx >= total) return;
  int n = idx % CNP;
  int r = idx / CNP;
  int c = r % CHD;
  int b = r / CHD;
  vt[idx] = (n < CN) ? (_Float16)pv[(b * CN + n) * CHD + c] : (_Float16)0.f;
}

// ---------------- relative-position bias precompute ----------------
__global__ void rel_hw_kernel(const float* __restrict__ q,
                              const float* __restrict__ rel_h,
                              const float* __restrict__ rel_w,
                              float* __restrict__ relhq,
                              float* __restrict__ relwq, int total) {
  int idx = blockIdx.x * blockDim.x + threadIdx.x;
  if (idx >= total) return;
  int k = idx % CH;
  int r = idx / CH;
  int w = r % CW; r /= CW;
  int h = r % CH; r /= CH;
  int t = r % CT;
  int b = r / CT;
  const float* qp = q + (b * CN + t * CHW1 + 1 + h * CW + w) * CHD;
  const float* rh = rel_h + (h - k + CH - 1) * CHD;
  const float* rw = rel_w + (w - k + CW - 1) * CHD;
  float sh = 0.f, sw = 0.f;
  for (int c = 0; c < CHD; ++c) {
    float qv = qp[c];
    sh += qv * rh[c];
    sw += qv * rw[c];
  }
  relhq[idx] = sh;
  relwq[idx] = sw;
}

__global__ void rel_t_kernel(const float* __restrict__ q,
                             const float* __restrict__ rel_t,
                             float* __restrict__ reltq, int total) {
  int idx = blockIdx.x * blockDim.x + threadIdx.x;
  if (idx >= total) return;
  int k = idx % CT;
  int r = idx / CT;
  int j = r % CHW1; r /= CHW1;
  int t = r % CT;
  int b = r / CT;
  const float* qp = q + (b * CN + t * CHW1 + j) * CHD;
  const float* rt = rel_t + (t - k + CT - 1) * CHD;
  float s = 0.f;
  for (int c = 0; c < CHD; ++c) s += qp[c] * rt[c];
  reltq[idx] = s;
}

// ---------------- fused flash-style attention ----------------
// One wave owns a 16-query tile; K tiles staged to LDS with async b128 copies;
// online softmax in registers; O accumulated via WMMA; no NxN buffer.
__global__ __launch_bounds__(128) void attention_kernel(
    const _Float16* __restrict__ qh, const _Float16* __restrict__ kh,
    const _Float16* __restrict__ vt, const float* __restrict__ qres,
    const float* __restrict__ relh, const float* __restrict__ relw,
    const float* __restrict__ relt, float* __restrict__ out) {
  __shared__ __align__(16) _Float16 ktile[4][32 * CHD];   // 6KB per wave
  __shared__ __align__(16) _Float16 pstage[4][16 * 32];   // P transpose tile
  int wid = threadIdx.x >> 5;
  int lane = threadIdx.x & 31;
  int tile = blockIdx.x * 4 + wid;
  if (tile >= CB * CQT) return;                 // wave-uniform
  int b = tile / CQT;
  int qt = tile - b * CQT;
  int q0 = qt * 16;

  const _Float16* qhB = qh + b * (CN * CHD);
  const _Float16* khB = kh + b * (CN * CHD);
  const _Float16* vtB = vt + b * (CHD * CNP);
  const float* qptr = qres + b * (CN * CHD);

  // preload scaled Q A-fragments (K = 96 -> 3 chunks of 32)
  v16h qf[3];
#pragma unroll
  for (int kc = 0; kc < 3; ++kc)
    qf[kc] = load_a_f16(qhB, CHD, q0, kc * 32, CN - 1);

  int mrow = (lane >> 4) << 3;  // row offset of this lane-half in D layout
  int body[8];
  int reltb[8], spb[8];
#pragma unroll
  for (int i = 0; i < 8; ++i) {
    int qi = q0 + mrow + i;
    if (qi > CN - 1) qi = CN - 1;
    int tq = qi / CHW1, jq = qi - tq * CHW1;
    reltb[i] = ((b * CT + tq) * CHW1 + jq) * CT;
    if (jq > 0) {
      int hq = (jq - 1) / CW, wq = (jq - 1) % CW;
      spb[i] = ((b * CT + tq) * CH + hq) * CW + wq;
      body[i] = 1;
    } else {
      spb[i] = 0;
      body[i] = 0;
    }
  }

  float mi[8], li[8];
  v8f o[6];
#pragma unroll
  for (int i = 0; i < 8; ++i) { mi[i] = -3.0e38f; li[i] = 0.f; }
#pragma unroll
  for (int f = 0; f < 6; ++f)
#pragma unroll
    for (int i = 0; i < 8; ++i) o[f][i] = 0.f;

  _Float16* kt = ktile[wid];
  uint32_t ldsbase = (uint32_t)(size_t)kt;

  for (int kt0 = 0; kt0 < CKT; ++kt0) {
    int k0 = kt0 * 32;
    // prefetch next K tile toward L2 (global_prefetch_b8)
    {
      int nxt = k0 + 32 + lane;
      if (nxt > CN - 1) nxt = CN - 1;
      __builtin_prefetch(khB + nxt * CHD, 0, 1);
    }
    // ---- stage 32x96 f16 K tile into per-wave LDS via async b128 copies ----
    // 32 keys * 192B = 6KB = 384 x 16B units; 32 lanes -> 12 issues.
#pragma unroll
    for (int it = 0; it < 12; ++it) {
      int elt = it * 32 + lane;          // 0..383
      int key = elt / 12, seg = elt % 12;
      int gk = k0 + key;
      if (gk > CN - 1) gk = CN - 1;
      const _Float16* gp = khB + (gk * CHD + seg * 8);
      uint32_t lva = ldsbase + (uint32_t)(elt * 16);
      asm volatile("global_load_async_to_lds_b128 %0, %1, off"
                   :: "v"(lva), "v"(gp) : "memory");
    }
    asm volatile("s_wait_asynccnt 0x0" ::: "memory");

    // ---- S = (scale*Q) @ K^T : two 16x16 halves over 32 keys ----
    v8f s0, s1;
#pragma unroll
    for (int i = 0; i < 8; ++i) { s0[i] = 0.f; s1[i] = 0.f; }
#pragma unroll
    for (int kc = 0; kc < 3; ++kc) {
      v16h kf0 = load_b_kf16_lds(kt, 0, kc * 32);
      v16h kf1 = load_b_kf16_lds(kt, 16, kc * 32);
      s0 = __builtin_amdgcn_wmma_f32_16x16x32_f16(false, qf[kc], false, kf0,
                                                  (short)0, s0, false, false);
      s1 = __builtin_amdgcn_wmma_f32_16x16x32_f16(false, qf[kc], false, kf1,
                                                  (short)0, s1, false, false);
    }
    // ---- decomposed rel-pos bias; mask OOB keys ----
#pragma unroll
    for (int h2 = 0; h2 < 2; ++h2) {
      int n = k0 + h2 * 16 + (lane & 15);
      if (n >= CN) {
#pragma unroll
        for (int i = 0; i < 8; ++i) {
          if (h2 == 0) s0[i] = -1.0e30f; else s1[i] = -1.0e30f;
        }
      } else {
        int tk = n / CHW1, jk = n - tk * CHW1;
        int hk = 0, wk = 0, kbody = 0;
        if (jk > 0) { hk = (jk - 1) / CW; wk = (jk - 1) % CW; kbody = 1; }
#pragma unroll
        for (int i = 0; i < 8; ++i) {
          float bias = relt[reltb[i] + tk];
          if (body[i] && kbody)
            bias += relh[spb[i] * CH + hk] + relw[spb[i] * CW + wk];
          if (h2 == 0) s0[i] += bias; else s1[i] += bias;
        }
      }
    }
    // ---- online softmax update (row spread over 16 lanes of each half) ----
#pragma unroll
    for (int i = 0; i < 8; ++i) {
      float rv = fmaxf(s0[i], s1[i]);
#pragma unroll
      for (int off = 1; off < 16; off <<= 1) rv = fmaxf(rv, __shfl_xor(rv, off, 32));
      float mnew = fmaxf(mi[i], rv);
      float corr = __expf(mi[i] - mnew);
      float p0 = __expf(s0[i] - mnew);
      float p1 = __expf(s1[i] - mnew);
      float rs = p0 + p1;
#pragma unroll
      for (int off = 1; off < 16; off <<= 1) rs += __shfl_xor(rs, off, 32);
      li[i] = li[i] * corr + rs;
      mi[i] = mnew;
      s0[i] = p0;
      s1[i] = p1;
#pragma unroll
      for (int f = 0; f < 6; ++f) o[f][i] *= corr;
    }
    // ---- transpose P (D layout -> A layout) through per-wave LDS tile ----
    _Float16* pp = pstage[wid];
    int col = lane & 15;
#pragma unroll
    for (int i = 0; i < 8; ++i) {
      pp[(mrow + i) * 32 + col]      = (_Float16)s0[i];
      pp[(mrow + i) * 32 + 16 + col] = (_Float16)s1[i];
    }
    asm volatile("s_wait_dscnt 0" ::: "memory");  // same-wave LDS RAW
    v16h pf;
    {
      int r = lane & 15;
      int kbq = (lane >> 4) << 3;
      v8h lo = *(const v8h*)&pp[r * 32 + kbq];
      v8h hi = *(const v8h*)&pp[r * 32 + kbq + 16];
#pragma unroll
      for (int i = 0; i < 8; ++i) { pf[i] = lo[i]; pf[8 + i] = hi[i]; }
    }
    // ---- O += P @ V : V^T fragments are contiguous b128 loads ----
#pragma unroll
    for (int f = 0; f < 6; ++f) {
      v16h vf = load_b_vt_f16(vtB, f * 16, k0);
      o[f] = __builtin_amdgcn_wmma_f32_16x16x32_f16(false, pf, false, vf,
                                                    (short)0, o[f], false, false);
    }
  }

  // epilogue: O/l + residual q (fp32); full tiles take unguarded fast path
  float* ob = out + b * (CN * CHD);
  int col = lane & 15;
  if (q0 + 16 <= CN) {
#pragma unroll
    for (int f = 0; f < 6; ++f) {
      float* p = ob + ((q0 + mrow) * CHD + f * 16 + col);
      const float* rq = qptr + ((q0 + mrow) * CHD + f * 16 + col);
#pragma unroll
      for (int i = 0; i < 8; ++i)
        p[i * CHD] = o[f][i] / li[i] + rq[i * CHD];
    }
  } else {
#pragma unroll
    for (int f = 0; f < 6; ++f) {
#pragma unroll
      for (int i = 0; i < 8; ++i) {
        int qi = q0 + mrow + i;
        if (qi < CN) {
          int c = f * 16 + col;
          ob[qi * CHD + c] = o[f][i] / li[i] + qptr[qi * CHD + c];
        }
      }
    }
  }
}

// ---------------- host launch ----------------
extern "C" void kernel_launch(void* const* d_in, const int* in_sizes, int n_in,
                              void* d_out, int out_size, void* d_ws, size_t ws_size,
                              hipStream_t stream) {
  const float* x        = (const float*)d_in[0];
  const float* qkv_w    = (const float*)d_in[1];
  const float* qkv_b    = (const float*)d_in[2];
  const float* pool_q_w = (const float*)d_in[3];
  const float* nq_g     = (const float*)d_in[4];
  const float* nq_b     = (const float*)d_in[5];
  const float* pool_k_w = (const float*)d_in[6];
  const float* nk_g     = (const float*)d_in[7];
  const float* nk_b     = (const float*)d_in[8];
  const float* pool_v_w = (const float*)d_in[9];
  const float* nv_g     = (const float*)d_in[10];
  const float* nv_b     = (const float*)d_in[11];
  const float* rel_h    = (const float*)d_in[12];
  const float* rel_w    = (const float*)d_in[13];
  const float* rel_t    = (const float*)d_in[14];
  const float* proj_w   = (const float*)d_in[15];
  const float* proj_b   = (const float*)d_in[16];

  float* ws = (float*)d_ws;
  size_t off = 0;
  float* qkv = ws + off; off += (size_t)CM * 288;
  float* pq  = ws + off; off += (size_t)CM * CHD;
  float* pk  = ws + off; off += (size_t)CM * CHD;
  float* pv  = ws + off; off += (size_t)CM * CHD;
  const int relhw_total = CB * CT * CH * CW * CH;      // 351232
  float* relhq = ws + off; off += relhw_total;
  float* relwq = ws + off; off += relhw_total;
  const int relt_total = CB * CT * CHW1 * CT;          // 100480
  float* reltq = ws + off; off += relt_total;
  float* attn_out = ws + off; off += (size_t)CM * CHD;
  // f16 region (16B aligned: float count so far is divisible by 4)
  _Float16* qh = (_Float16*)(ws + off);
  _Float16* kh = qh + (size_t)CM * CHD;
  _Float16* vt = kh + (size_t)CM * CHD;                // CB*CHD*CNP halves

  // 1) QKV GEMM: (12560 x 96) @ (96 x 288) + bias
  {
    int tiles = (CM / 16) * (288 / 16);
    wmma_gemm_bias<<<(tiles + 3) / 4, 128, 0, stream>>>(x, qkv_w, qkv_b, qkv,
                                                        CM, 288, 96);
  }
  // 2) depthwise pool + LN for q/k/v
  {
    int total = CM * CHD;
    int blk = (total + 255) / 256;
    pool_conv<<<blk, 256, 0, stream>>>(qkv + 0, 288, pool_q_w, pq, total);
    pool_conv<<<blk, 256, 0, stream>>>(qkv + 96, 288, pool_k_w, pk, total);
    pool_conv<<<blk, 256, 0, stream>>>(qkv + 192, 288, pool_v_w, pv, total);
    int tb = (CM + 255) / 256;
    layernorm_inplace<<<tb, 256, 0, stream>>>(pq, nq_g, nq_b, CM);
    layernorm_inplace<<<tb, 256, 0, stream>>>(pk, nk_g, nk_b, CM);
    layernorm_inplace<<<tb, 256, 0, stream>>>(pv, nv_g, nv_b, CM);
  }
  // 3) rel-pos bias tables + f16 repack (Q scaled, K, V^T padded)
  rel_hw_kernel<<<(relhw_total + 255) / 256, 256, 0, stream>>>(
      pq, rel_h, rel_w, relhq, relwq, relhw_total);
  rel_t_kernel<<<(relt_total + 255) / 256, 256, 0, stream>>>(
      pq, rel_t, reltq, relt_total);
  {
    float scale = 1.0f / sqrtf((float)CHD);
    int total = CM * CHD;
    cvt_qk<<<(total + 255) / 256, 256, 0, stream>>>(pq, pk, qh, kh, scale, total);
    int vtot = CB * CHD * CNP;
    cvt_vt<<<(vtot + 255) / 256, 256, 0, stream>>>(pv, vt, vtot);
  }
  // 4) fused attention (async K staging, softmax + rel bias + @V + residual)
  {
    int tiles = CB * CQT;  // 786 wave-tiles
    attention_kernel<<<(tiles + 3) / 4, 128, 0, stream>>>(
        qh, kh, vt, pq, relhq, relwq, reltq, attn_out);
  }
  // 5) output projection: (12560 x 96) @ (96 x 96) + bias -> d_out (fp32)
  {
    int tiles = (CM / 16) * (96 / 16);
    wmma_gemm_bias<<<(tiles + 3) / 4, 128, 0, stream>>>(attn_out, proj_w, proj_b,
                                                        (float*)d_out, CM, 96, 96);
  }
}
